// ParallelMLayer_33122787787613
// MI455X (gfx1250) — compile-verified, matrix operand order checked by reading, not run on hardware
//
#include <hip/hip_runtime.h>
#include <hip/hip_bf16.h>
#include <math.h>

// ---------------------------------------------------------------------------
// Types for CDNA5 WMMA (wave32): 16x16x32 bf16 -> f32
// ---------------------------------------------------------------------------
typedef __attribute__((ext_vector_type(16))) __bf16 v16bf;
typedef __attribute__((ext_vector_type(8)))  __bf16 bf16x8;
typedef __attribute__((ext_vector_type(8)))  float  v8f;

// Problem constants (match reference)
#define BSZ   4
#define LSEQ  2048
#define DDIM  2048
#define MROWS (BSZ * LSEQ)      // 8192
#define NCOLS DDIM              // 2048
#define KDIM  DDIM              // 2048
#define NCHUNK 16
#define CHUNKL (LSEQ / NCHUNK)  // 128
#define BASE_DT 0.1f

// LDS tile geometry: K-step 64, padded row stride 72 bf16 (144 B, 16B-aligned)
#define TKSTEP 64
#define LDSPAD 72

// ---------------------------------------------------------------------------
// CDNA5 async memory->LDS copy (VGLOBAL GLOBAL_LOAD_ASYNC_TO_LDS_B128,
// tracked by ASYNCcnt). VDST is the per-lane LDS byte address; per ISA 10.2
// the LDS aperture truncates the generic address to addr[31:0], so the low
// 32 bits of a generic __shared__ pointer are the LDS address.
// ---------------------------------------------------------------------------
__device__ __forceinline__ void async_copy_b128(void* lds_dst,
                                                const void* gsrc) {
  unsigned l = (unsigned)(unsigned long long)lds_dst;
  asm volatile("global_load_async_to_lds_b128 %0, %1, off"
               :
               : "v"(l), "v"(gsrc)
               : "memory");
}

__device__ __forceinline__ void wait_async0() {
  asm volatile("s_wait_asynccnt 0" ::: "memory");
}

// ---------------------------------------------------------------------------
// 1) f32 -> bf16 elementwise convert (force). Coalesced streaming pass.
//    After this, force_bf16 (32MB) + 3x W_bf16 (24MB) fit in the 192MB L2,
//    so every GEMM tile re-read is an L2 hit, not HBM.
// ---------------------------------------------------------------------------
__global__ void k_f32_to_bf16(const float* __restrict__ src,
                              __bf16* __restrict__ dst, int n) {
  int i = blockIdx.x * blockDim.x + threadIdx.x;
  if (i < n) dst[i] = (__bf16)src[i];
}

// ---------------------------------------------------------------------------
// 2) Convert + transpose weights: WT[z][n][k] = (bf16) W[z][k][n]
//    LDS-tiled so both the read and the write are coalesced.
//    z: 0 = WA, 1 = Wdt, 2 = WB  (matches GEMM epilogue selection)
// ---------------------------------------------------------------------------
__global__ void k_convert_transpose_w(const float* __restrict__ WA,
                                      const float* __restrict__ Wdt,
                                      const float* __restrict__ WB,
                                      __bf16* __restrict__ WT) {
  __shared__ float tile[32][33];
  const int z = blockIdx.z;
  const float* W = (z == 0) ? WA : (z == 1) ? Wdt : WB;
  __bf16* dst = WT + (size_t)z * NCOLS * KDIM;
  const int tx = threadIdx.x;           // 0..31
  const int ty = threadIdx.y;           // 0..7
  const int n0 = blockIdx.x * 32;
  const int k0 = blockIdx.y * 32;
#pragma unroll
  for (int s = 0; s < 32; s += 8)
    tile[ty + s][tx] = W[(size_t)(k0 + ty + s) * NCOLS + (n0 + tx)];
  __syncthreads();
#pragma unroll
  for (int s = 0; s < 32; s += 8)
    dst[(size_t)(n0 + ty + s) * KDIM + (k0 + tx)] = (__bf16)tile[tx][ty + s];
}

// ---------------------------------------------------------------------------
// 3) Fused triple GEMM with bf16 WMMA, async-LDS double-buffered pipeline.
//    Block = 256 threads = 8 waves; output tile 128x128; K stepped by 64.
//    Wave grid 2(m) x 4(n): each wave owns 64x32 = 4x2 WMMA accumulators,
//    16 v_wmma per staged tile. Next tile's async copies are issued before
//    computing on the current tile: one s_wait_asynccnt + one barrier per
//    64-K step.
//    gridDim.z selects weight/epilogue:
//      z=0: A   = sigmoid(force@WA + bA)
//      z=1: dt  = softplus(force@Wdt + bdt) * BASE_DT * scales
//      z=2: Braw= force@WB + bB            (dt applied later in the scan)
// ---------------------------------------------------------------------------
__global__ __launch_bounds__(256)
void k_gemm_bf16_wmma(const __bf16* __restrict__ Fb,
                      const __bf16* __restrict__ WTall,
                      float* __restrict__ Aout,
                      float* __restrict__ dtOut,
                      float* __restrict__ BrawOut,
                      const float* __restrict__ bA,
                      const float* __restrict__ bdt,
                      const float* __restrict__ bB,
                      const float* __restrict__ scales) {
  // Double-buffered padded LDS tiles: 2 x 128 x 72 bf16 each = 36 KB/matrix
  __shared__ __align__(16) __bf16 As[2][128][LDSPAD];
  __shared__ __align__(16) __bf16 Bs[2][128][LDSPAD];

  const int z  = blockIdx.z;
  const __bf16* WT = WTall + (size_t)z * NCOLS * KDIM;
  const int m0 = blockIdx.y * 128;
  const int n0 = blockIdx.x * 128;

  const int tid  = threadIdx.x;
  const int lane = tid & 31;
  const int wid  = tid >> 5;
  const int wm   = wid >> 2;      // 0..1  (64-row slab)
  const int wn   = wid & 3;       // 0..3  (32-col slab)

  // staging assignment: thread -> (row, 32-element half of the 64-wide row)
  const int srow  = tid >> 1;     // 0..127
  const int shalf = (tid & 1) * 32;

  // Issue the 8 async b128 copies staging one 128x64 tile of A and B^T.
  auto stage = [&](int buf, int k0) {
    const __bf16* gA = Fb + (size_t)(m0 + srow) * KDIM + k0 + shalf;
    const __bf16* gB = WT + (size_t)(n0 + srow) * KDIM + k0 + shalf;
#pragma unroll
    for (int c = 0; c < 4; ++c) {
      async_copy_b128(&As[buf][srow][shalf + c * 8], gA + c * 8);
      async_copy_b128(&Bs[buf][srow][shalf + c * 8], gB + c * 8);
    }
  };

  v8f acc[4][2] = {};

  // Fragment addressing per the 16-bit A/B VGPR layout:
  // lane 0-15 : rows m, K = {kb..kb+7, kb+16..kb+23}, kb = 0
  // lane 16-31: same rows,                            kb = 8
  const int kb = (lane >> 4) * 8;
  const int r  = lane & 15;

  // ---- prologue: stage tile 0, wait, publish ----
  stage(0, 0);
  wait_async0();
  __syncthreads();

  const int nIter = KDIM / TKSTEP;   // 32
  for (int it = 0; it < nIter; ++it) {
    const int cur = it & 1;
    const int nxt = cur ^ 1;

    // Kick off next tile's async copies; they overlap the WMMAs below.
    // (Buffer `nxt` was last read in iteration it-1, released by that
    //  iteration's trailing barrier.)
    if (it + 1 < nIter) stage(nxt, (it + 1) * TKSTEP);

    // ---- compute on the current staged 64-K tile: 2 x (12 ds_b128 + 8 wmma)
#pragma unroll
    for (int ks = 0; ks < TKSTEP; ks += 32) {
      v16bf aF[4], bF[2];
#pragma unroll
      for (int i = 0; i < 4; ++i) {
        bf16x8 lo = *(const bf16x8*)&As[cur][wm * 64 + i * 16 + r][ks + kb];
        bf16x8 hi = *(const bf16x8*)&As[cur][wm * 64 + i * 16 + r][ks + kb + 16];
        aF[i] = __builtin_shufflevector(lo, hi, 0, 1, 2, 3, 4, 5, 6, 7, 8, 9,
                                        10, 11, 12, 13, 14, 15);
      }
#pragma unroll
      for (int j = 0; j < 2; ++j) {
        bf16x8 lo = *(const bf16x8*)&Bs[cur][wn * 32 + j * 16 + r][ks + kb];
        bf16x8 hi = *(const bf16x8*)&Bs[cur][wn * 32 + j * 16 + r][ks + kb + 16];
        bF[j] = __builtin_shufflevector(lo, hi, 0, 1, 2, 3, 4, 5, 6, 7, 8, 9,
                                        10, 11, 12, 13, 14, 15);
      }
#pragma unroll
      for (int i = 0; i < 4; ++i)
#pragma unroll
        for (int j = 0; j < 2; ++j)
          acc[i][j] = __builtin_amdgcn_wmma_f32_16x16x32_bf16(
              false, aF[i], false, bF[j], (short)0, acc[i][j], false, false);
    }

    // Own async copies done + everyone past reads of `cur` -> publish `nxt`,
    // release `cur` for restaging.
    wait_async0();
    __syncthreads();
  }

  // ---- epilogue: C/D layout -> lane l, vgpr e: row = e + (l>>4)*8, col = l&15
  float* outPtr      = (z == 0) ? Aout : (z == 1) ? dtOut : BrawOut;
  const float* bias  = (z == 0) ? bA   : (z == 1) ? bdt   : bB;
  const int colBase  = n0 + wn * 32 + (lane & 15);
  const int rowHalf  = (lane >> 4) * 8;

#pragma unroll
  for (int i = 0; i < 4; ++i) {
    const int rBase = m0 + wm * 64 + i * 16 + rowHalf;
#pragma unroll
    for (int j = 0; j < 2; ++j) {
      const int col    = colBase + j * 16;
      const float bcol = bias[col];
      const float scol = scales[col];
#pragma unroll
      for (int e = 0; e < 8; ++e) {
        float x = acc[i][j][e] + bcol;
        float res;
        if (z == 0)      res = 1.0f / (1.0f + __expf(-x));
        else if (z == 1) res = (x > 20.0f ? x : log1pf(__expf(x))) * BASE_DT * scol;
        else             res = x;
        outPtr[(size_t)(rBase + e) * NCOLS + col] = res;
      }
    }
  }
}

// ---------------------------------------------------------------------------
// 4) Chunked associative scan, pass 1: per-chunk aggregates.
//    For chunk-local recurrence q_t = a_t q_{t-1} + b_t (q_in = 0),
//    P_t = prod a, we store (P_end, q_end, sum P_t*dt_t, sum q_t*dt_t).
// ---------------------------------------------------------------------------
__global__ void k_scan_pass1(const float* __restrict__ Af,
                             const float* __restrict__ dtf,
                             const float* __restrict__ Braw,
                             float4* __restrict__ agg) {
  const int tid = blockIdx.x * blockDim.x + threadIdx.x;
  if (tid >= BSZ * NCHUNK * DDIM) return;
  const int d  = tid % DDIM;
  const int rc = tid / DDIM;       // b*NCHUNK + c
  const int c  = rc % NCHUNK;
  const int b  = rc / NCHUNK;
  size_t base = ((size_t)(b * LSEQ + c * CHUNKL)) * DDIM + d;
  float P = 1.0f, q = 0.0f, sP = 0.0f, sq = 0.0f;
  for (int t = 0; t < CHUNKL; ++t, base += DDIM) {
    const float a   = Af[base];
    const float dtv = dtf[base];
    const float bv  = Braw[base] * dtv;
    q  = a * q + bv;
    P *= a;
    sP += P * dtv;
    sq += q * dtv;
  }
  agg[(size_t)rc * DDIM + d] = make_float4(P, q, sP, sq);
}

// ---------------------------------------------------------------------------
// 5) Pass 2: sequential scan over the NCHUNK aggregates per (b, d) to get
//    the v-carry and cumsum-carry entering each chunk.
// ---------------------------------------------------------------------------
__global__ void k_scan_pass2(const float4* __restrict__ agg,
                             float2* __restrict__ carry) {
  const int tid = blockIdx.x * blockDim.x + threadIdx.x;
  if (tid >= BSZ * DDIM) return;
  const int d = tid % DDIM;
  const int b = tid / DDIM;
  float v = 0.0f, x = 0.0f;
  for (int c = 0; c < NCHUNK; ++c) {
    const size_t idx = ((size_t)(b * NCHUNK + c)) * DDIM + d;
    const float4 f = agg[idx];
    carry[idx] = make_float2(v, x);
    x += v * f.z + f.w;          // chunk's contribution to cumsum(v*dt)
    v  = f.x * v + f.y;          // v leaving the chunk
  }
}

// ---------------------------------------------------------------------------
// 6) Pass 3: replay each chunk with its carries, write v_seq and x_seq.
// ---------------------------------------------------------------------------
__global__ void k_scan_pass3(const float* __restrict__ Af,
                             const float* __restrict__ dtf,
                             const float* __restrict__ Braw,
                             const float2* __restrict__ carry,
                             float* __restrict__ xOut,
                             float* __restrict__ vOut) {
  const int tid = blockIdx.x * blockDim.x + threadIdx.x;
  if (tid >= BSZ * NCHUNK * DDIM) return;
  const int d  = tid % DDIM;
  const int rc = tid / DDIM;
  const int c  = rc % NCHUNK;
  const int b  = rc / NCHUNK;
  const float2 cv = carry[(size_t)rc * DDIM + d];
  float v = cv.x, x = cv.y;
  size_t base = ((size_t)(b * LSEQ + c * CHUNKL)) * DDIM + d;
  for (int t = 0; t < CHUNKL; ++t, base += DDIM) {
    const float a   = Af[base];
    const float dtv = dtf[base];
    const float bv  = Braw[base] * dtv;
    v = a * v + bv;
    vOut[base] = v;
    x += v * dtv;
    xOut[base] = x;
  }
}

// ---------------------------------------------------------------------------
// Host launcher
// Inputs (setup_inputs order): 0:x 1:v 2:force 3:WA 4:bA 5:WB 6:bB 7:Wdt
//                              8:bdt 9:scales
// Output: concat(x_seq, v_seq), each B*L*D f32.
// ---------------------------------------------------------------------------
extern "C" void kernel_launch(void* const* d_in, const int* in_sizes, int n_in,
                              void* d_out, int out_size, void* d_ws,
                              size_t ws_size, hipStream_t stream) {
  const float* force  = (const float*)d_in[2];
  const float* WA     = (const float*)d_in[3];
  const float* bA     = (const float*)d_in[4];
  const float* WB     = (const float*)d_in[5];
  const float* bB     = (const float*)d_in[6];
  const float* Wdt    = (const float*)d_in[7];
  const float* bdt    = (const float*)d_in[8];
  const float* scales = (const float*)d_in[9];

  char* ws = (char*)d_ws;
  size_t off = 0;
  __bf16* Fb = (__bf16*)(ws + off);  off += (size_t)MROWS * KDIM * sizeof(__bf16);
  __bf16* WT = (__bf16*)(ws + off);  off += 3ull * NCOLS * KDIM * sizeof(__bf16);
  float* Af   = (float*)(ws + off);  off += (size_t)MROWS * NCOLS * sizeof(float);
  float* dtf  = (float*)(ws + off);  off += (size_t)MROWS * NCOLS * sizeof(float);
  float* Braw = (float*)(ws + off);  off += (size_t)MROWS * NCOLS * sizeof(float);
  float4* agg   = (float4*)(ws + off); off += (size_t)BSZ * NCHUNK * DDIM * sizeof(float4);
  float2* carry = (float2*)(ws + off); off += (size_t)BSZ * NCHUNK * DDIM * sizeof(float2);

  float* xOut = (float*)d_out;
  float* vOut = xOut + (size_t)MROWS * NCOLS;   // B*L*D

  // 1) force f32 -> bf16 (resident in L2 for all GEMM tiles)
  {
    const int n = MROWS * KDIM;
    k_f32_to_bf16<<<(n + 255) / 256, 256, 0, stream>>>(force, Fb, n);
  }
  // 2) weights f32 -> bf16, transposed (coalesced staging in the GEMM)
  {
    dim3 blk(32, 8, 1), grd(NCOLS / 32, KDIM / 32, 3);
    k_convert_transpose_w<<<grd, blk, 0, stream>>>(WA, Wdt, WB, WT);
  }
  // 3) fused triple GEMM + activations (WMMA bf16 -> f32, async-LDS pipeline)
  {
    dim3 grd(NCOLS / 128, MROWS / 128, 3);
    k_gemm_bf16_wmma<<<grd, 256, 0, stream>>>(Fb, WT, Af, dtf, Braw, bA, bdt,
                                              bB, scales);
  }
  // 4-6) chunked linear-recurrence scan + cumsum
  {
    const int n1 = BSZ * NCHUNK * DDIM;
    k_scan_pass1<<<(n1 + 255) / 256, 256, 0, stream>>>(Af, dtf, Braw, agg);
    const int n2 = BSZ * DDIM;
    k_scan_pass2<<<(n2 + 255) / 256, 256, 0, stream>>>(agg, carry);
    k_scan_pass3<<<(n1 + 255) / 256, 256, 0, stream>>>(Af, dtf, Braw, carry,
                                                       xOut, vOut);
  }
}
